// DriftBrake_37546604101721
// MI455X (gfx1250) — matrix operand order repaired and checked by reading, not compiled, not run on hardware
//
#include <hip/hip_runtime.h>
#include <cstddef>
#include <cstdint>

// Problem constants (match reference)
#define BB 8
#define NN 8192
#define SS 1024
#define KK 16
#define MROWS (BB * SS * KK)   // 131072 grouped rows per SA stage
#define EPSF 1e-5f

// ---------------------------------------------------------------- WMMA types
typedef __attribute__((ext_vector_type(16))) __bf16 v16bf;
typedef __attribute__((ext_vector_type(8)))  __bf16 v8bf;   // 16B, natural align 16
typedef __attribute__((ext_vector_type(8)))  float  v8f;

union BF16Frag { v16bf v; v8bf h[2]; };

__device__ __forceinline__ unsigned short f2bf(float x) {
  unsigned int b = __float_as_uint(x);
  b += 0x7FFFu + ((b >> 16) & 1u);   // round-to-nearest-even to bf16
  return (unsigned short)(b >> 16);
}
__device__ __forceinline__ unsigned int pk_bf16(float lo, float hi) {
  return (unsigned int)f2bf(lo) | ((unsigned int)f2bf(hi) << 16);
}

// ---------------------------------------------------------------- init
// Zero BN stats + padded scale/shift buffers; build identity affine.
__global__ void init_ws_kernel(float* stats, int nstats,
                               float* idscale, float* idshift, int nid) {
  int t = blockIdx.x * blockDim.x + threadIdx.x;
  if (t < nstats) stats[t] = 0.f;
  if (t < nid) { idscale[t] = 1.f; idshift[t] = 0.f; }
}

// ---------------------------------------------------------------- transpose [B,3,n] -> [B,n,3]
__global__ void transpose_bcn_kernel(const float* __restrict__ in,
                                     float* __restrict__ out, int n, int total) {
  int t = blockIdx.x * blockDim.x + threadIdx.x;
  if (t >= total) return;
  int i = t % n; int b = t / n;
  const float* src = in + (size_t)b * 3 * n + i;
  float* dst = out + (size_t)t * 3;
  dst[0] = src[0];
  dst[1] = src[(size_t)n];
  dst[2] = src[(size_t)2 * n];
}

// ---------------------------------------------------------------- farthest point sampling
// One workgroup (1024 threads = 32 wave32) per batch; dist array lives in LDS.
__global__ void __launch_bounds__(1024)
fps_kernel(const float* __restrict__ xyz, int n, int npoint,
           int* __restrict__ idx, float* __restrict__ newxyz) {
  __shared__ float dist[NN];
  __shared__ float rv[32];
  __shared__ int   ri[32];
  __shared__ int   sfar;
  int b = blockIdx.x;
  int tid = threadIdx.x;
  const int T = blockDim.x;
  const float* px = xyz + (size_t)b * n * 3;
  for (int i = tid; i < n; i += T) dist[i] = 1e10f;
  if (tid == 0) sfar = 0;
  __syncthreads();
  for (int s = 0; s < npoint; ++s) {
    int far = sfar;
    if (tid == 0) idx[b * npoint + s] = far;
    if (tid < 3) newxyz[((size_t)b * npoint + s) * 3 + tid] = px[(size_t)far * 3 + tid];
    float cx = px[far * 3 + 0], cy = px[far * 3 + 1], cz = px[far * 3 + 2];
    float bv = -1.f; int bidx = 0;
    for (int i = tid; i < n; i += T) {
      float dx = px[i * 3 + 0] - cx;
      float dy = px[i * 3 + 1] - cy;
      float dz = px[i * 3 + 2] - cz;
      float d = dx * dx + dy * dy + dz * dz;
      float nd = fminf(dist[i], d);
      dist[i] = nd;
      if (nd > bv || (nd == bv && i < bidx)) { bv = nd; bidx = i; }
    }
    for (int off = 16; off > 0; off >>= 1) {        // wave32 argmax reduce
      float ov = __shfl_xor(bv, off, 32);
      int   oi = __shfl_xor(bidx, off, 32);
      if (ov > bv || (ov == bv && oi < bidx)) { bv = ov; bidx = oi; }
    }
    int wid = tid >> 5;
    if ((tid & 31) == 0) { rv[wid] = bv; ri[wid] = bidx; }
    __syncthreads();
    if (wid == 0) {
      float v2 = rv[tid & 31]; int i2 = ri[tid & 31];  // exactly 32 waves
      for (int off = 16; off > 0; off >>= 1) {
        float ov = __shfl_xor(v2, off, 32);
        int   oi = __shfl_xor(i2, off, 32);
        if (ov > v2 || (ov == v2 && oi < i2)) { v2 = ov; i2 = oi; }
      }
      if (tid == 0) sfar = i2;
    }
    __syncthreads();
  }
}

// ---------------------------------------------------------------- KNN (k=16)
// One wave per centroid; per-lane register-sorted top-16 then LDS merge.
__global__ void __launch_bounds__(256)
knn_kernel(const float* __restrict__ xyz, const float* __restrict__ cent,
           int n, int* __restrict__ knn_out) {
  __shared__ float md[8][512];
  __shared__ int   mi[8][512];
  int wid  = threadIdx.x >> 5;
  int lane = threadIdx.x & 31;
  int wave = blockIdx.x * 8 + wid;          // wave in [0, B*S)
  int b = wave >> 10;
  const float* px = xyz + (size_t)b * n * 3;
  const float* pc = cent + (size_t)wave * 3;
  float cx = pc[0], cy = pc[1], cz = pc[2];
  float bd[16]; int bi[16];
#pragma unroll
  for (int j = 0; j < 16; ++j) { bd[j] = 1e30f; bi[j] = 0; }
  for (int i = lane; i < n; i += 32) {
    float dx = px[i * 3 + 0] - cx;
    float dy = px[i * 3 + 1] - cy;
    float dz = px[i * 3 + 2] - cz;
    float d = dx * dx + dy * dy + dz * dz;
    if (d < bd[15]) {
      bd[15] = d; bi[15] = i;
#pragma unroll
      for (int j = 15; j > 0; --j) {
        if (bd[j] < bd[j - 1]) {
          float tf = bd[j]; bd[j] = bd[j - 1]; bd[j - 1] = tf;
          int   ti = bi[j]; bi[j] = bi[j - 1]; bi[j - 1] = ti;
        }
      }
    }
  }
#pragma unroll
  for (int j = 0; j < 16; ++j) {
    md[wid][lane * 16 + j] = bd[j];
    mi[wid][lane * 16 + j] = bi[j];
  }
  __syncthreads();
  for (int k = 0; k < 16; ++k) {
    float mv = 1e31f; int mp = 0;
#pragma unroll
    for (int j = 0; j < 16; ++j) {
      float v = md[wid][lane * 16 + j];
      if (v < mv) { mv = v; mp = lane * 16 + j; }
    }
    for (int off = 16; off > 0; off >>= 1) {
      float ov = __shfl_xor(mv, off, 32);
      int   op = __shfl_xor(mp, off, 32);
      if (ov < mv || (ov == mv && op < mp)) { mv = ov; mp = op; }
    }
    if (lane == 0) {
      knn_out[(size_t)wave * 16 + k] = mi[wid][mp];
      md[wid][mp] = 1e31f;
    }
    __syncthreads();
  }
}

// ---------------------------------------------------------------- grouping + concat (emit bf16)
// Row t = (b,s,k): [rel_xyz(3) | feat(Cf) | zero-pad ... ldo]
__global__ void group_concat_kernel(const float* __restrict__ xyz, int n,
                                    const float* __restrict__ cent,
                                    const int* __restrict__ knn,
                                    const float* __restrict__ feats, int Cf,
                                    unsigned short* __restrict__ Outb, int ldo, int total) {
  int t = blockIdx.x * blockDim.x + threadIdx.x;
  if (t >= total) return;
  int b = t >> 14;                 // /(S*K)
  int j = knn[t];
  unsigned short* row = Outb + (size_t)t * ldo;
  const float* pj = xyz + ((size_t)b * n + j) * 3;
  const float* pc = cent + (size_t)(t >> 4) * 3;   // (b*S+s)
  row[0] = f2bf(pj[0] - pc[0]);
  row[1] = f2bf(pj[1] - pc[1]);
  row[2] = f2bf(pj[2] - pc[2]);
  const float* pf = feats + ((size_t)b * n + j) * Cf;
  for (int c = 0; c < Cf; ++c) row[3 + c] = f2bf(pf[c]);
  for (int c = 3 + Cf; c < ldo; ++c) row[c] = 0;
}

// ---------------------------------------------------------------- weight pre-pack
// Pack W [Cout,Cin] (f32) into bf16 B-fragments:
//   Wb[((tileN*ktiles + kt)*32 + lane)*16 + j] = W[col][kg], col=tileN*16+(lane&15),
//   kg = kt*32 + (lane>>4)*16 + j   (zero outside Cin/Cout)
__global__ void pack_weights_kernel(const float* __restrict__ W, int Cin, int Cout,
                                    int ktiles, int ntiles,
                                    unsigned short* __restrict__ Wb) {
  int t = blockIdx.x * blockDim.x + threadIdx.x;
  int total = ntiles * ktiles * 512;
  if (t >= total) return;
  int j = t & 15;
  int lane = (t >> 4) & 31;
  int r2 = t >> 9;
  int kt = r2 % ktiles;
  int tileN = r2 / ktiles;
  int col = tileN * 16 + (lane & 15);
  int kg = kt * 32 + (lane >> 4) * 16 + j;
  float w = (col < Cout && kg < Cin) ? W[(size_t)col * Cin + kg] : 0.f;
  Wb[t] = f2bf(w);
}

// ---------------------------------------------------------------- activation prep
// y = relu(x*scale[c] + shift[c]) -> bf16, 8 elements/thread, fully vectorized.
// lda is a power of two (mask = lda-1); scale/shift are lda-padded (zeros in pad).
__global__ void prep_act_kernel(const float* __restrict__ In,
                                const float* __restrict__ scale,
                                const float* __restrict__ shift,
                                int cmask, int relu,
                                unsigned short* __restrict__ Outb, int total8) {
  int t = blockIdx.x * blockDim.x + threadIdx.x;
  if (t >= total8) return;
  size_t base = (size_t)t * 8;
  int c0 = (int)(base & (size_t)cmask);
  float4 x0 = *(const float4*)(In + base);
  float4 x1 = *(const float4*)(In + base + 4);
  float4 s0 = *(const float4*)(scale + c0);
  float4 s1 = *(const float4*)(scale + c0 + 4);
  float4 h0 = *(const float4*)(shift + c0);
  float4 h1 = *(const float4*)(shift + c0 + 4);
  float y0 = x0.x * s0.x + h0.x, y1 = x0.y * s0.y + h0.y;
  float y2 = x0.z * s0.z + h0.z, y3 = x0.w * s0.w + h0.w;
  float y4 = x1.x * s1.x + h1.x, y5 = x1.y * s1.y + h1.y;
  float y6 = x1.z * s1.z + h1.z, y7 = x1.w * s1.w + h1.w;
  if (relu) {
    y0 = fmaxf(y0, 0.f); y1 = fmaxf(y1, 0.f); y2 = fmaxf(y2, 0.f); y3 = fmaxf(y3, 0.f);
    y4 = fmaxf(y4, 0.f); y5 = fmaxf(y5, 0.f); y6 = fmaxf(y6, 0.f); y7 = fmaxf(y7, 0.f);
  }
  uint4 o;
  o.x = pk_bf16(y0, y1); o.y = pk_bf16(y2, y3);
  o.z = pk_bf16(y4, y5); o.w = pk_bf16(y6, y7);
  *(uint4*)(Outb + base) = o;
}

// ---------------------------------------------------------------- WMMA GEMM (bf16 in, f32 out + stats)
// One wave per 16x32 output tile (two N-tiles share one A fragment).
// Pure b128 fragment loads; Cout is a multiple of 32 here (ntiles even).
__global__ void __launch_bounds__(256)
gemm_wmma_kernel(const unsigned short* __restrict__ Abf, int lda,
                 const unsigned short* __restrict__ Wb,
                 const float* __restrict__ bias,
                 float* __restrict__ Out, int ldo,
                 float* __restrict__ ssum, float* __restrict__ ssqs,
                 int mtiles, int ntiles2) {
  int wave = (blockIdx.x * blockDim.x + threadIdx.x) >> 5;
  if (wave >= mtiles * ntiles2) return;         // wave-uniform: EXEC stays all-1s
  int tnp   = wave % ntiles2;                   // N-tile pair index
  int tileM = wave / ntiles2;
  int lane = threadIdx.x & 31;
  int half = lane >> 4;
  int lq = lane & 15;
  int ktiles = lda >> 5;                        // lda is a multiple of 32
  int row = tileM * 16 + lq;

  const unsigned short* ap  = Abf + (size_t)row * lda + half * 16;
  const unsigned short* wp0 = Wb + (((size_t)(2 * tnp) * ktiles) * 32 + lane) * 16;
  const unsigned short* wp1 = wp0 + (size_t)ktiles * 512;

  v8f acc0 = {0.f, 0.f, 0.f, 0.f, 0.f, 0.f, 0.f, 0.f};
  v8f acc1 = {0.f, 0.f, 0.f, 0.f, 0.f, 0.f, 0.f, 0.f};
  for (int kt = 0; kt < ktiles; ++kt) {
    BF16Frag af, b0, b1;
    af.h[0] = *(const v8bf*)(ap);         // K {0..7} / {16..23}
    af.h[1] = *(const v8bf*)(ap + 8);     // K {8..15} / {24..31}
    b0.h[0] = *(const v8bf*)(wp0);
    b0.h[1] = *(const v8bf*)(wp0 + 8);
    b1.h[0] = *(const v8bf*)(wp1);
    b1.h[1] = *(const v8bf*)(wp1 + 8);
    __builtin_prefetch(ap + 32, 0, 1);    // speculative: dropped if past end
    acc0 = __builtin_amdgcn_wmma_f32_16x16x32_bf16(
        false, af.v, false, b0.v, (short)0, acc0, false, false);
    acc1 = __builtin_amdgcn_wmma_f32_16x16x32_bf16(
        false, af.v, false, b1.v, (short)0, acc1, false, false);
    ap  += 32;
    wp0 += 512;
    wp1 += 512;
  }

  int col0 = tnp * 32 + lq;
  int col1 = col0 + 16;
  float bc0 = (bias != nullptr) ? bias[col0] : 0.f;
  float bc1 = (bias != nullptr) ? bias[col1] : 0.f;
  float lsum0 = 0.f, lsq0 = 0.f, lsum1 = 0.f, lsq1 = 0.f;
  int m0 = tileM * 16 + half * 8;
  float* orow = Out + (size_t)m0 * ldo;
#pragma unroll
  for (int v = 0; v < 8; ++v) {
    float v0 = acc0[v] + bc0;
    float v1 = acc1[v] + bc1;
    orow[col0] = v0;
    orow[col1] = v1;
    orow += ldo;
    lsum0 += v0; lsq0 += v0 * v0;
    lsum1 += v1; lsq1 += v1 * v1;
  }
  lsum0 += __shfl_xor(lsum0, 16, 32);   // lanes l and l^16 hold the same channel
  lsq0  += __shfl_xor(lsq0, 16, 32);
  lsum1 += __shfl_xor(lsum1, 16, 32);
  lsq1  += __shfl_xor(lsq1, 16, 32);
  if (ssum != nullptr && half == 0) {
    atomicAdd(&ssum[col0], lsum0);
    atomicAdd(&ssqs[col0], lsq0);
    atomicAdd(&ssum[col1], lsum1);
    atomicAdd(&ssqs[col1], lsq1);
  }
}

// ---------------------------------------------------------------- BN finalize: stats -> scale/shift
__global__ void finalize_bn_kernel(const float* __restrict__ ssum,
                                   const float* __restrict__ ssqs,
                                   const float* __restrict__ gamma,
                                   const float* __restrict__ beta,
                                   float invM, int C,
                                   float* __restrict__ scale,
                                   float* __restrict__ shift) {
  int c = blockIdx.x * blockDim.x + threadIdx.x;
  if (c >= C) return;
  float mean = ssum[c] * invM;
  float var = ssqs[c] * invM - mean * mean;
  float rstd = rsqrtf(var + EPSF);
  float sc = gamma[c] * rstd;
  scale[c] = sc;
  shift[c] = beta[c] - mean * sc;
}

// ---------------------------------------------------------------- BN + ReLU + max-pool over K
__global__ void bn_relu_maxpool_kernel(const float* __restrict__ In, int ldi, int C,
                                       const float* __restrict__ scale,
                                       const float* __restrict__ shift,
                                       float* __restrict__ Outp, int total) {
  int t = blockIdx.x * blockDim.x + threadIdx.x;
  if (t >= total) return;
  int c = t % C; int bs = t / C;
  float sc = scale[c], sh = shift[c];
  float m = -1e30f;
  const float* p = In + (size_t)bs * KK * ldi + c;
  for (int k = 0; k < KK; ++k) {
    float v = fmaxf(p[(size_t)k * ldi] * sc + sh, 0.f);
    m = fmaxf(m, v);
  }
  Outp[(size_t)bs * C + c] = m;
}

// ---------------------------------------------------------------- final mean over points
__global__ void final_reduce_kernel(const float* __restrict__ Q, float* __restrict__ out) {
  int t = blockIdx.x * blockDim.x + threadIdx.x;   // 2048 threads
  if (t >= BB * 256) return;
  int p = t & 255; int b = t >> 8;
  const float* q = Q + (size_t)b * SS * 256 + p;
  float s = 0.f;
  for (int i = 0; i < SS; ++i) s += q[(size_t)i * 256];
  out[t] = s / 8192.f;
}

// ================================================================ host launch
extern "C" void kernel_launch(void* const* d_in, const int* in_sizes, int n_in,
                              void* d_out, int out_size, void* d_ws, size_t ws_size,
                              hipStream_t stream) {
  (void)in_sizes; (void)n_in; (void)out_size; (void)ws_size;
  const float* pc       = (const float*)d_in[0];
  const float* feature  = (const float*)d_in[1];
  const float* sa1_w[3] = {(const float*)d_in[2],  (const float*)d_in[3],  (const float*)d_in[4]};
  const float* sa1_b[3] = {(const float*)d_in[5],  (const float*)d_in[6],  (const float*)d_in[7]};
  const float* sa1_g[3] = {(const float*)d_in[8],  (const float*)d_in[9],  (const float*)d_in[10]};
  const float* sa1_t[3] = {(const float*)d_in[11], (const float*)d_in[12], (const float*)d_in[13]};
  const float* sa2_w[3] = {(const float*)d_in[14], (const float*)d_in[15], (const float*)d_in[16]};
  const float* sa2_b[3] = {(const float*)d_in[17], (const float*)d_in[18], (const float*)d_in[19]};
  const float* sa2_g[3] = {(const float*)d_in[20], (const float*)d_in[21], (const float*)d_in[22]};
  const float* sa2_t[3] = {(const float*)d_in[23], (const float*)d_in[24], (const float*)d_in[25]};
  const float* conv1_w  = (const float*)d_in[26];
  const float* bn1_g    = (const float*)d_in[27];
  const float* bn1_t    = (const float*)d_in[28];
  const float* conv2_w  = (const float*)d_in[29];
  const float* conv2_b  = (const float*)d_in[30];
  float* out = (float*)d_out;

  // Workspace carve-out (256B aligned)
  char* ws = (char*)d_ws;
  size_t off = 0;
  auto alloc = [&](size_t bytes) -> void* {
    void* p = (void*)(ws + off);
    off += (bytes + 255) & ~(size_t)255;
    return p;
  };
  float* xyz1  = (float*)alloc((size_t)BB * NN * 3 * 4);
  float* pts1  = (float*)alloc((size_t)BB * NN * 3 * 4);
  int*   fps1  = (int*)  alloc((size_t)BB * SS * 4);
  float* nx1   = (float*)alloc((size_t)BB * SS * 3 * 4);
  int*   knn1  = (int*)  alloc((size_t)BB * SS * KK * 4);
  int*   fps2  = (int*)  alloc((size_t)BB * SS * 4);
  float* nx2   = (float*)alloc((size_t)BB * SS * 3 * 4);
  int*   knn2  = (int*)  alloc((size_t)BB * SS * KK * 4);
  float* l1f   = (float*)alloc((size_t)BB * SS * 64 * 4);
  float* l2f   = (float*)alloc((size_t)BB * SS * 128 * 4);
  // stats(16*256) + bnscale(8*256) + bnshift(8*256) are contiguous: 8192 floats
  float* stats = (float*)alloc((size_t)16 * 256 * 4);
  float* bnsc  = (float*)alloc((size_t)8 * 256 * 4);
  float* bnsh  = (float*)alloc((size_t)8 * 256 * 4);
  float* idsc  = (float*)alloc((size_t)256 * 4);
  float* idsh  = (float*)alloc((size_t)256 * 4);
  // packed bf16 weights per layer: ntiles*ktiles*512 u16
  static const int L_nt[8] = {2, 2, 4, 4, 4, 8, 16, 16};
  static const int L_kt[8] = {1, 1, 1, 3, 2, 2, 4, 8};
  unsigned short* wb[8];
  for (int l = 0; l < 8; ++l)
    wb[l] = (unsigned short*)alloc((size_t)L_nt[l] * L_kt[l] * 512 * 2);
  float*          bufR = (float*)alloc((size_t)MROWS * 128 * 4);           // raw f32 out
  unsigned short* bfA  = (unsigned short*)alloc((size_t)MROWS * 128 * 2);  // bf16 acts
  unsigned short* bfB  = (unsigned short*)alloc((size_t)MROWS * 128 * 2);

  auto st = [&](int l) { return stats + (size_t)l * 512; };
  auto sq = [&](int l) { return stats + (size_t)l * 512 + 256; };
  auto sc = [&](int l) { return bnsc + (size_t)l * 256; };
  auto sh = [&](int l) { return bnsh + (size_t)l * 256; };

  auto pack = [&](int l, const float* W, int Cin) {
    int total = L_nt[l] * L_kt[l] * 512;
    pack_weights_kernel<<<(total + 255) / 256, 256, 0, stream>>>(
        W, Cin, L_nt[l] * 16, L_kt[l], L_nt[l], wb[l]);
  };
  auto gemm = [&](const unsigned short* Abf, int lda, int l, const float* bias,
                  float* O, int ldo, float* su, float* sqs, int M) {
    int mtiles = M / 16, ntiles2 = L_nt[l] / 2;   // two N-tiles per wave
    int blocks = (mtiles * ntiles2 + 7) / 8;
    gemm_wmma_kernel<<<blocks, 256, 0, stream>>>(Abf, lda, wb[l], bias, O, ldo,
                                                 su, sqs, mtiles, ntiles2);
  };
  auto finalize = [&](int l, const float* g, const float* bt, int C, float invM) {
    finalize_bn_kernel<<<1, 256, 0, stream>>>(st(l), sq(l), g, bt, invM, C, sc(l), sh(l));
  };
  auto prep = [&](const float* In, const float* s_, const float* h_, int lda, int relu,
                  unsigned short* Ob, size_t total) {
    int t8 = (int)(total / 8);
    prep_act_kernel<<<(t8 + 255) / 256, 256, 0, stream>>>(In, s_, h_, lda - 1, relu, Ob, t8);
  };

  const float invM1 = 1.f / (float)MROWS;
  const float invM2 = 1.f / (float)(BB * SS);

  // ---- init: zero stats + scale/shift pads (8192 contiguous floats), identity affine
  init_ws_kernel<<<32, 256, 0, stream>>>(stats, 32 * 256, idsc, idsh, 256);

  // ---- pack all weights (cheap, every call: graph-replay safe)
  pack(0, sa1_w[0], 6);   pack(1, sa1_w[1], 32);  pack(2, sa1_w[2], 32);
  pack(3, sa2_w[0], 67);  pack(4, sa2_w[1], 64);  pack(5, sa2_w[2], 64);
  pack(6, conv1_w, 128);  pack(7, conv2_w, 256);

  // ---- transpose inputs to [B,n,3]
  transpose_bcn_kernel<<<(BB * NN + 255) / 256, 256, 0, stream>>>(pc, xyz1, NN, BB * NN);
  transpose_bcn_kernel<<<(BB * NN + 255) / 256, 256, 0, stream>>>(feature, pts1, NN, BB * NN);

  // ---- SA1: FPS + KNN + group (bf16, K-pad 6->32 zeroed)
  fps_kernel<<<BB, 1024, 0, stream>>>(xyz1, NN, SS, fps1, nx1);
  knn_kernel<<<BB * SS / 8, 256, 0, stream>>>(xyz1, nx1, NN, knn1);
  group_concat_kernel<<<MROWS / 256, 256, 0, stream>>>(xyz1, NN, nx1, knn1, pts1, 3,
                                                       bfA, 32, MROWS);
  // ---- SA1 MLP (6->32->32->64)
  gemm(bfA, 32, 0, sa1_b[0], bufR, 32, st(0), sq(0), MROWS);
  finalize(0, sa1_g[0], sa1_t[0], 32, invM1);
  prep(bufR, sc(0), sh(0), 32, 1, bfB, (size_t)MROWS * 32);
  gemm(bfB, 32, 1, sa1_b[1], bufR, 32, st(1), sq(1), MROWS);
  finalize(1, sa1_g[1], sa1_t[1], 32, invM1);
  prep(bufR, sc(1), sh(1), 32, 1, bfA, (size_t)MROWS * 32);
  gemm(bfA, 32, 2, sa1_b[2], bufR, 64, st(2), sq(2), MROWS);
  finalize(2, sa1_g[2], sa1_t[2], 64, invM1);
  bn_relu_maxpool_kernel<<<(BB * SS * 64 + 255) / 256, 256, 0, stream>>>(
      bufR, 64, 64, sc(2), sh(2), l1f, BB * SS * 64);

  // ---- SA2: FPS + KNN + group (bf16, K-pad 67->96 zeroed)
  fps_kernel<<<BB, 1024, 0, stream>>>(nx1, SS, SS, fps2, nx2);
  knn_kernel<<<BB * SS / 8, 256, 0, stream>>>(nx1, nx2, SS, knn2);
  group_concat_kernel<<<MROWS / 256, 256, 0, stream>>>(nx1, SS, nx2, knn2, l1f, 64,
                                                       bfA, 96, MROWS);
  // ---- SA2 MLP (67->64->64->128)
  gemm(bfA, 96, 3, sa2_b[0], bufR, 64, st(3), sq(3), MROWS);
  finalize(3, sa2_g[0], sa2_t[0], 64, invM1);
  prep(bufR, sc(3), sh(3), 64, 1, bfB, (size_t)MROWS * 64);
  gemm(bfB, 64, 4, sa2_b[1], bufR, 64, st(4), sq(4), MROWS);
  finalize(4, sa2_g[1], sa2_t[1], 64, invM1);
  prep(bufR, sc(4), sh(4), 64, 1, bfA, (size_t)MROWS * 64);
  gemm(bfA, 64, 5, sa2_b[2], bufR, 128, st(5), sq(5), MROWS);
  finalize(5, sa2_g[2], sa2_t[2], 128, invM1);
  bn_relu_maxpool_kernel<<<(BB * SS * 128 + 255) / 256, 256, 0, stream>>>(
      bufR, 128, 128, sc(5), sh(5), l2f, BB * SS * 128);

  // ---- head: conv1 (no bias) + BN1 + ReLU + conv2 (+bias) + mean
  prep(l2f, idsc, idsh, 128, 0, bfA, (size_t)BB * SS * 128);
  gemm(bfA, 128, 6, nullptr, bufR, 256, st(6), sq(6), BB * SS);
  finalize(6, bn1_g, bn1_t, 256, invM2);
  prep(bufR, sc(6), sh(6), 256, 1, bfB, (size_t)BB * SS * 256);
  gemm(bfB, 256, 7, conv2_b, bufR, 256, nullptr, nullptr, BB * SS);
  final_reduce_kernel<<<(BB * 256 + 255) / 256, 256, 0, stream>>>(bufR, out);
}
